// CausalSelfAttention_63531156242542
// MI455X (gfx1250) — matrix operand order repaired and checked by reading, not compiled, not run on hardware
//
#include <hip/hip_runtime.h>
#include <hip/hip_bf16.h>

// ---------------------------------------------------------------------------
// CausalSelfAttention on MI455X (gfx1250), wave32 + v_wmma_f32_16x16x32_f16.
// B=2, T=2048, C=1024, H=16, d=64. fp32 in/out, f16 compute w/ fp32 accum.
// ---------------------------------------------------------------------------

#define B_  2
#define T_  2048
#define C_  1024
#define H_  16
#define D_  64
#define M_  (B_ * T_)          // 4096 rows in token-major GEMMs

typedef __attribute__((ext_vector_type(16))) _Float16 v16h;
typedef __attribute__((ext_vector_type(8)))  _Float16 v8h;
typedef __attribute__((ext_vector_type(4)))  _Float16 v4h;
typedef __attribute__((ext_vector_type(8)))  float    v8f;
typedef __attribute__((ext_vector_type(4)))  int      v4i;

union Frag16 { v16h v; v8h h8[2]; };

// ---- A fragment: 16x32 f16 (ISA 7.12.2 "16-bit A-Matrix 16x32") -----------
// lanes 0-15: row=lane, halves 0..7 = K0..7, halves 8..15 = K16..23
// lanes 16-31: row=lane-16, halves 0..7 = K8..15, halves 8..15 = K24..31
__device__ __forceinline__ v16h load_fragA(const _Float16* base, int ld) {
    int lane = threadIdx.x & 31;
    int r    = lane & 15;
    int kh   = (lane & 16) ? 8 : 0;
    const _Float16* p = base + r * ld + kh;
    Frag16 f;
    f.h8[0] = *(const v8h*)(p);
    f.h8[1] = *(const v8h*)(p + 16);
    return f.v;
}

// ---- B fragment: 32x16 f16 (per ISA 7.12.4 B-matrix pattern) ---------------
// lanes 0-15: col N=lane, halves 0..7 = K0..7, halves 8..15 = K16..23
// lanes 16-31: col N=lane-16, halves 0..7 = K8..15, halves 8..15 = K24..31
// Source tile stored [N][K] row-major with pitch `ld` halves.
__device__ __forceinline__ v16h load_fragB(const _Float16* base, int ld) {
    int lane = threadIdx.x & 31;
    int n    = lane & 15;
    int kh   = (lane & 16) ? 8 : 0;
    const _Float16* p = base + n * ld + kh;
    Frag16 f;
    f.h8[0] = *(const v8h*)(p);
    f.h8[1] = *(const v8h*)(p + 16);
    return f.v;
}

__device__ __forceinline__ v8f wmma_f16(v16h a, v16h b, v8f c) {
    return __builtin_amdgcn_wmma_f32_16x16x32_f16(
        /*neg_a=*/false, a, /*neg_b=*/false, b,
        /*c_mod=*/(short)0, c, /*reuse_a=*/false, /*reuse_b=*/false);
}

// ---- async global->LDS copy (16B per lane), gfx1250 ASYNCcnt path ----------
#if __has_builtin(__builtin_amdgcn_global_load_async_to_lds_b128)
#define HAS_ASYNC_LDS 1
#endif

__device__ __forceinline__ void copy16_g2l(const _Float16* g, _Float16* l) {
#ifdef HAS_ASYNC_LDS
    // prototype: (v4i addrspace(1)*, v4i addrspace(3)*, imm offset, imm cpol)
    __builtin_amdgcn_global_load_async_to_lds_b128(
        (__attribute__((address_space(1))) v4i*)(uintptr_t)g,
        (__attribute__((address_space(3))) v4i*)(unsigned int)(uintptr_t)l,
        0, 0);
#else
    *(v8h*)l = *(const v8h*)g;
#endif
}

__device__ __forceinline__ void async_wait0() {
#ifdef HAS_ASYNC_LDS
#if __has_builtin(__builtin_amdgcn_s_wait_asynccnt)
    __builtin_amdgcn_s_wait_asynccnt(0);
#else
    asm volatile("s_wait_asynccnt 0x0" ::: "memory");
#endif
#endif
}

// ---------------------------------------------------------------------------
// fp32 -> f16 conversion (vectorized x4)
// ---------------------------------------------------------------------------
__global__ void cvt_f32_to_f16(const float* __restrict__ in,
                               _Float16* __restrict__ out, int n) {
    int i = (blockIdx.x * blockDim.x + threadIdx.x) * 4;
    if (i < n) {
        float4 f = *(const float4*)&in[i];
        v4h o;
        o[0] = (_Float16)f.x; o[1] = (_Float16)f.y;
        o[2] = (_Float16)f.z; o[3] = (_Float16)f.w;
        *(v4h*)&out[i] = o;
    }
}

// ---------------------------------------------------------------------------
// QKV GEMM:  [4096 x 1024] @ [1024 x 1024] + bias -> f16 in [B,H,T,D] layout.
// gridDim.z selects {Q,K,V}. Q gets 1/sqrt(d) folded into the epilogue.
// Block 256 thr = 8 waves (4 along M x 2 along N); tile 128x64, K-step 32.
// ---------------------------------------------------------------------------
#define GQ_LDA 40   // 32 + 8 pad (halves); row pitch 80B (16B multiple)
__global__ __launch_bounds__(256) void gemm_qkv(
    const _Float16* __restrict__ Xh,
    const _Float16* __restrict__ Wqh, const _Float16* __restrict__ Wkh,
    const _Float16* __restrict__ Wvh,
    const float* __restrict__ bq, const float* __restrict__ bk,
    const float* __restrict__ bv,
    _Float16* __restrict__ Qh, _Float16* __restrict__ Kh,
    _Float16* __restrict__ Vh) {
    __shared__ _Float16 As[128 * GQ_LDA];
    __shared__ _Float16 Bs[64 * GQ_LDA];   // stored transposed: [n][k]

    const int z = blockIdx.z;
    const _Float16* W  = (z == 0) ? Wqh : (z == 1) ? Wkh : Wvh;
    const float*  bias = (z == 0) ? bq  : (z == 1) ? bk  : bv;
    _Float16*     Out  = (z == 0) ? Qh  : (z == 1) ? Kh  : Vh;
    const float  scale = (z == 0) ? 0.125f : 1.0f;   // 1/sqrt(64)

    const int m0  = blockIdx.y * 128;
    const int n0  = blockIdx.x * 64;
    const int tid = threadIdx.x;
    const int wave = tid >> 5;
    const int wm = wave & 3, wn = wave >> 2;

    v8f acc[2][2];
    #pragma unroll
    for (int i = 0; i < 2; ++i)
        #pragma unroll
        for (int j = 0; j < 2; ++j)
            #pragma unroll
            for (int r = 0; r < 8; ++r) acc[i][j][r] = 0.0f;

    for (int k0 = 0; k0 < C_; k0 += 32) {
        // A tile: 128x32 halves, async 16B per thread x2
        #pragma unroll
        for (int it = 0; it < 2; ++it) {
            int idx = (tid + it * 256) * 8;
            int r = idx >> 5, c = idx & 31;
            copy16_g2l(&Xh[(m0 + r) * C_ + k0 + c], &As[r * GQ_LDA + c]);
            if (k0 + 32 < C_)
                __builtin_prefetch(&Xh[(m0 + r) * C_ + k0 + 32 + c], 0, 3);
        }
        // B tile: 32x64, coalesced read, transposed store -> Bs[n][k]
        {
            int idx = tid * 8;
            int kr = idx >> 6, nc = idx & 63;
            v8h w = *(const v8h*)&W[(k0 + kr) * C_ + n0 + nc];
            #pragma unroll
            for (int e = 0; e < 8; ++e) Bs[(nc + e) * GQ_LDA + kr] = w[e];
        }
        async_wait0();
        __syncthreads();

        v16h a0 = load_fragA(&As[(wm * 32 +  0) * GQ_LDA], GQ_LDA);
        v16h a1 = load_fragA(&As[(wm * 32 + 16) * GQ_LDA], GQ_LDA);
        v16h b0 = load_fragB(&Bs[(wn * 32 +  0) * GQ_LDA], GQ_LDA);
        v16h b1 = load_fragB(&Bs[(wn * 32 + 16) * GQ_LDA], GQ_LDA);
        acc[0][0] = wmma_f16(a0, b0, acc[0][0]);
        acc[0][1] = wmma_f16(a0, b1, acc[0][1]);
        acc[1][0] = wmma_f16(a1, b0, acc[1][0]);
        acc[1][1] = wmma_f16(a1, b1, acc[1][1]);
        __syncthreads();
    }

    // epilogue: C layout -> [B,H,T,D] f16
    const int lane = tid & 31;
    const int cnn  = lane & 15;
    const int cm   = (lane >> 4) << 3;
    #pragma unroll
    for (int i = 0; i < 2; ++i) {
        #pragma unroll
        for (int j = 0; j < 2; ++j) {
            int n = n0 + wn * 32 + j * 16 + cnn;
            float bn = bias[n];
            int hh = n >> 6, dd = n & 63;
            #pragma unroll
            for (int r = 0; r < 8; ++r) {
                int m  = m0 + wm * 32 + i * 16 + cm + r;
                int bb = m >> 11;           // m / T_
                int t  = m & (T_ - 1);
                float v = (acc[i][j][r] + bn) * scale;
                Out[(((size_t)(bb * H_ + hh) * T_) + t) * D_ + dd] = (_Float16)v;
            }
        }
    }
}

// ---------------------------------------------------------------------------
// Flash attention: block = 128 thr (4 waves), 64 query rows / block,
// one (b,h) per blockIdx.y. Online softmax in WMMA accumulator layout.
// ---------------------------------------------------------------------------
#define K_LD 72     // 64 + 8 pad (halves), row pitch 144B
#define V_LD 40     // 32 + 8 pad
#define P_LD 40
#define NEGI (-1.0e30f)

__global__ __launch_bounds__(128) void flash_attn(
    const _Float16* __restrict__ Qh, const _Float16* __restrict__ Kh,
    const _Float16* __restrict__ Vh, _Float16* __restrict__ Yh) {
    __shared__ _Float16 Ks[32 * K_LD];        // [key][d]   (B-frag src: N=key rows, K=d contiguous)
    __shared__ _Float16 Vs[64 * V_LD];        // [d][key]   (B-frag src: N=d rows, K=key contiguous)
    __shared__ _Float16 Ps[4 * 16 * P_LD];    // per-wave P staging 16x32

    const int bh = blockIdx.y;                // b*H + h
    const int b  = bh >> 4, h = bh & 15;
    const int q0 = blockIdx.x * 64;
    const int tid = threadIdx.x;
    const int wave = tid >> 5, lane = tid & 31;
    const int qw = q0 + wave * 16;            // this wave's 16 query rows
    const int cnn = lane & 15;
    const int cm  = (lane >> 4) << 3;

    const _Float16* Qb = Qh + (size_t)bh * T_ * D_;
    const _Float16* Kb = Kh + (size_t)bh * T_ * D_;
    const _Float16* Vb = Vh + (size_t)bh * T_ * D_;

    // Q fragments straight from global ([t][d] row-major == A layout source)
    v16h qa0 = load_fragA(Qb + (size_t)qw * D_ +  0, D_);
    v16h qa1 = load_fragA(Qb + (size_t)qw * D_ + 32, D_);

    v8f accO[4];
    float mrun[8], lrun[8];
    #pragma unroll
    for (int n = 0; n < 4; ++n)
        #pragma unroll
        for (int r = 0; r < 8; ++r) accO[n][r] = 0.0f;
    #pragma unroll
    for (int r = 0; r < 8; ++r) { mrun[r] = NEGI; lrun[r] = 0.0f; }

    for (int k0 = 0; k0 < q0 + 64; k0 += 32) {
        // stage K tile [32][64] as-is (async global->LDS)
        #pragma unroll
        for (int it = 0; it < 2; ++it) {
            int idx = (tid + it * 128) * 8;
            int r = idx >> 6, c = idx & 63;
            copy16_g2l(&Kb[(size_t)(k0 + r) * D_ + c], &Ks[r * K_LD + c]);
        }
        // stage V tile transposed -> Vs[d][key] (needs VGPR transpose)
        #pragma unroll
        for (int it = 0; it < 2; ++it) {
            int idx = (tid + it * 128) * 8;
            int r = idx >> 6, c = idx & 63;
            v8h vv = *(const v8h*)&Vb[(size_t)(k0 + r) * D_ + c];
            #pragma unroll
            for (int e = 0; e < 8; ++e) Vs[(c + e) * V_LD + r] = vv[e];
        }
        async_wait0();
        __syncthreads();

        // S = Q K^T for two 16-key sub-tiles
        float s[2][8];
        #pragma unroll
        for (int j = 0; j < 2; ++j) {
            v16h b0 = load_fragB(&Ks[(j * 16) * K_LD +  0], K_LD);
            v16h b1 = load_fragB(&Ks[(j * 16) * K_LD + 32], K_LD);
            v8f sa;
            #pragma unroll
            for (int r = 0; r < 8; ++r) sa[r] = 0.0f;
            sa = wmma_f16(qa0, b0, sa);
            sa = wmma_f16(qa1, b1, sa);
            int ki = k0 + j * 16 + cnn;
            #pragma unroll
            for (int r = 0; r < 8; ++r) {
                int qi = qw + cm + r;
                s[j][r] = (ki <= qi) ? sa[r] : NEGI;
            }
        }

        // online softmax; row M=cm+r lives in a 16-lane half-wave (N=lane&15)
        #pragma unroll
        for (int r = 0; r < 8; ++r) {
            float mx = fmaxf(s[0][r], s[1][r]);
            #pragma unroll
            for (int msk = 1; msk < 16; msk <<= 1)
                mx = fmaxf(mx, __shfl_xor(mx, msk, 32));
            float newm = fmaxf(mrun[r], mx);
            float p0 = 0.0f, p1 = 0.0f, alpha = 1.0f;
            if (newm > -1.0e29f) {
                p0 = (s[0][r] > -1.0e29f) ? __expf(s[0][r] - newm) : 0.0f;
                p1 = (s[1][r] > -1.0e29f) ? __expf(s[1][r] - newm) : 0.0f;
                alpha = (mrun[r] > -1.0e29f) ? __expf(mrun[r] - newm) : 0.0f;
            }
            float rs = p0 + p1;
            #pragma unroll
            for (int msk = 1; msk < 16; msk <<= 1)
                rs += __shfl_xor(rs, msk, 32);
            lrun[r] = lrun[r] * alpha + rs;
            mrun[r] = newm;
            #pragma unroll
            for (int n = 0; n < 4; ++n) accO[n][r] *= alpha;
            int prow = cm + r;
            Ps[(wave * 16 + prow) * P_LD +      cnn] = (_Float16)p0;
            Ps[(wave * 16 + prow) * P_LD + 16 + cnn] = (_Float16)p1;
        }

        // O += P V   (LDS ops from same wave are in-order: no barrier needed)
        v16h pa = load_fragA(&Ps[(wave * 16) * P_LD], P_LD);
        #pragma unroll
        for (int n = 0; n < 4; ++n) {
            v16h vbf = load_fragB(&Vs[(n * 16) * V_LD], V_LD);
            accO[n] = wmma_f16(pa, vbf, accO[n]);
        }
        __syncthreads();
    }

    // normalize and write Y as f16 [B,T,C] for the output projection
    #pragma unroll
    for (int r = 0; r < 8; ++r) {
        float inv = (lrun[r] > 0.0f) ? (1.0f / lrun[r]) : 0.0f;
        int t = qw + cm + r;
        #pragma unroll
        for (int n = 0; n < 4; ++n) {
            Yh[((size_t)b * T_ + t) * C_ + h * D_ + n * 16 + cnn] =
                (_Float16)(accO[n][r] * inv);
        }
    }
}

// ---------------------------------------------------------------------------
// Output projection: Yh[4096x1024] @ Wp[1024x1024] + bp -> fp32 d_out
// ---------------------------------------------------------------------------
__global__ __launch_bounds__(256) void gemm_proj(
    const _Float16* __restrict__ Ah, const _Float16* __restrict__ Wh,
    const float* __restrict__ bias, float* __restrict__ Out) {
    __shared__ _Float16 As[128 * GQ_LDA];
    __shared__ _Float16 Bs[64 * GQ_LDA];

    const int m0  = blockIdx.y * 128;
    const int n0  = blockIdx.x * 64;
    const int tid = threadIdx.x;
    const int wave = tid >> 5;
    const int wm = wave & 3, wn = wave >> 2;

    v8f acc[2][2];
    #pragma unroll
    for (int i = 0; i < 2; ++i)
        #pragma unroll
        for (int j = 0; j < 2; ++j)
            #pragma unroll
            for (int r = 0; r < 8; ++r) acc[i][j][r] = 0.0f;

    for (int k0 = 0; k0 < C_; k0 += 32) {
        #pragma unroll
        for (int it = 0; it < 2; ++it) {
            int idx = (tid + it * 256) * 8;
            int r = idx >> 5, c = idx & 31;
            copy16_g2l(&Ah[(m0 + r) * C_ + k0 + c], &As[r * GQ_LDA + c]);
            if (k0 + 32 < C_)
                __builtin_prefetch(&Ah[(m0 + r) * C_ + k0 + 32 + c], 0, 3);
        }
        {
            int idx = tid * 8;
            int kr = idx >> 6, nc = idx & 63;
            v8h w = *(const v8h*)&Wh[(k0 + kr) * C_ + n0 + nc];
            #pragma unroll
            for (int e = 0; e < 8; ++e) Bs[(nc + e) * GQ_LDA + kr] = w[e];
        }
        async_wait0();
        __syncthreads();

        v16h a0 = load_fragA(&As[(wm * 32 +  0) * GQ_LDA], GQ_LDA);
        v16h a1 = load_fragA(&As[(wm * 32 + 16) * GQ_LDA], GQ_LDA);
        v16h b0 = load_fragB(&Bs[(wn * 32 +  0) * GQ_LDA], GQ_LDA);
        v16h b1 = load_fragB(&Bs[(wn * 32 + 16) * GQ_LDA], GQ_LDA);
        acc[0][0] = wmma_f16(a0, b0, acc[0][0]);
        acc[0][1] = wmma_f16(a0, b1, acc[0][1]);
        acc[1][0] = wmma_f16(a1, b0, acc[1][0]);
        acc[1][1] = wmma_f16(a1, b1, acc[1][1]);
        __syncthreads();
    }

    const int lane = tid & 31;
    const int cnn  = lane & 15;
    const int cm   = (lane >> 4) << 3;
    #pragma unroll
    for (int i = 0; i < 2; ++i) {
        #pragma unroll
        for (int j = 0; j < 2; ++j) {
            int n = n0 + wn * 32 + j * 16 + cnn;
            float bn = bias[n];
            #pragma unroll
            for (int r = 0; r < 8; ++r) {
                int m = m0 + wm * 32 + i * 16 + cm + r;
                Out[(size_t)m * C_ + n] = acc[i][j][r] + bn;
            }
        }
    }
}

// ---------------------------------------------------------------------------
// launch
// ---------------------------------------------------------------------------
extern "C" void kernel_launch(void* const* d_in, const int* in_sizes, int n_in,
                              void* d_out, int out_size, void* d_ws, size_t ws_size,
                              hipStream_t stream) {
    const float* X  = (const float*)d_in[0];
    const float* Wq = (const float*)d_in[1];
    const float* bq = (const float*)d_in[2];
    const float* Wk = (const float*)d_in[3];
    const float* bk = (const float*)d_in[4];
    const float* Wv = (const float*)d_in[5];
    const float* bv = (const float*)d_in[6];
    const float* Wp = (const float*)d_in[7];
    const float* bp = (const float*)d_in[8];

    char* ws = (char*)d_ws;
    const size_t nX = (size_t)M_ * C_;      // 4096*1024
    const size_t nW = (size_t)C_ * C_;      // 1024*1024
    _Float16* Xh  = (_Float16*)ws;  ws += nX * 2;
    _Float16* Wqh = (_Float16*)ws;  ws += nW * 2;
    _Float16* Wkh = (_Float16*)ws;  ws += nW * 2;
    _Float16* Wvh = (_Float16*)ws;  ws += nW * 2;
    _Float16* Wph = (_Float16*)ws;  ws += nW * 2;
    _Float16* Qh  = (_Float16*)ws;  ws += nX * 2;
    _Float16* Kh  = (_Float16*)ws;  ws += nX * 2;
    _Float16* Vh  = (_Float16*)ws;  ws += nX * 2;
    _Float16* Yh  = (_Float16*)ws;  ws += nX * 2;

    // 1) f32 -> f16 casts
    cvt_f32_to_f16<<<(int)(nX / 4 / 256), 256, 0, stream>>>(X,  Xh,  (int)nX);
    cvt_f32_to_f16<<<(int)(nW / 4 / 256), 256, 0, stream>>>(Wq, Wqh, (int)nW);
    cvt_f32_to_f16<<<(int)(nW / 4 / 256), 256, 0, stream>>>(Wk, Wkh, (int)nW);
    cvt_f32_to_f16<<<(int)(nW / 4 / 256), 256, 0, stream>>>(Wv, Wvh, (int)nW);
    cvt_f32_to_f16<<<(int)(nW / 4 / 256), 256, 0, stream>>>(Wp, Wph, (int)nW);

    // 2) fused QKV projection (z = 0/1/2 -> Q/K/V), Q pre-scaled by 1/sqrt(d)
    gemm_qkv<<<dim3(C_ / 64, M_ / 128, 3), 256, 0, stream>>>(
        Xh, Wqh, Wkh, Wvh, bq, bk, bv, Qh, Kh, Vh);

    // 3) causal flash attention: grid (T/64 query tiles, B*H heads)
    flash_attn<<<dim3(T_ / 64, B_ * H_), 128, 0, stream>>>(Qh, Kh, Vh, Yh);

    // 4) output projection -> fp32 d_out
    gemm_proj<<<dim3(C_ / 64, M_ / 128), 256, 0, stream>>>(
        Yh, Wph, bp, (float*)d_out);
}
